// GroupedQueryAttention_76991583748666
// MI455X (gfx1250) — compile-verified, matrix operand order checked by reading, not run on hardware
//
#include <hip/hip_runtime.h>

// ---------------- CDNA5 (gfx1250) wave32 WMMA types ----------------
typedef __attribute__((ext_vector_type(16))) __bf16 v16bf;
typedef __attribute__((ext_vector_type(8)))  float  v8f;

#define BATCH    4
#define SEQ      2048
#define IN_DIM   1024
#define KV_HEADS 4
#define HEAD_DIM 64
#define GROUPS   4
#define QKV_OUT  768   // 256 (group-summed Q) + 256 K + 256 V
#define O_DIM    256   // KV_HEADS * HEAD_DIM

#if __has_builtin(__builtin_amdgcn_global_load_async_to_lds_b128)
#define HAVE_ASYNC_LDS 1
typedef __attribute__((__vector_size__(4 * sizeof(int)))) int v4i;
typedef __attribute__((address_space(1))) v4i as1_v4i;   // prints as __device__
typedef __attribute__((address_space(3))) v4i as3_v4i;   // prints as __shared__
#else
#define HAVE_ASYNC_LDS 0
#endif

__device__ __forceinline__ unsigned short f2bf(float f) {
  unsigned u = __builtin_bit_cast(unsigned, f);
  u += 0x7FFFu + ((u >> 16) & 1u);   // round-to-nearest-even
  return (unsigned short)(u >> 16);
}

union FragBF {                // 16 bf16 = 8 VGPRs (A or B fragment)
  v16bf v;
  uint4 q[2];
  unsigned short h[16];
};

__device__ __forceinline__ v8f wmma_bf16(v16bf a, v16bf b, v8f c) {
  // D = A(16x32) * B(32x16) + C, fp32 accumulate
  return __builtin_amdgcn_wmma_f32_16x16x32_bf16(
      /*neg_a=*/false, a, /*neg_b=*/false, b,
      /*c_mod=*/(short)0, c, /*reuse_a=*/false, /*reuse_b=*/false);
}

// 16B copy global -> LDS: async (ASYNCcnt path) or sync fallback.
__device__ __forceinline__ void stage16(const unsigned short* gsrc,
                                        unsigned short* ldst) {
#if HAVE_ASYNC_LDS
  __builtin_amdgcn_global_load_async_to_lds_b128(
      (as1_v4i*)(unsigned long long)gsrc,
      (as3_v4i*)(unsigned)(unsigned long long)ldst, 0, 0);
#else
  *(uint4*)ldst = *(const uint4*)gsrc;
#endif
}

// ------------------------------------------------------------------
// Kernel A: build bf16 weights in B-fragment-friendly [out][k] layout.
//   Wqkv_t rows 0..255   : sum_g Wq[:, kv*4+g, d]   (effective Q weight)
//   Wqkv_t rows 256..511 : Wk ; rows 512..767 : Wv
//   Wo_t  [1024][256]    : Wo transposed
// ------------------------------------------------------------------
__global__ void __launch_bounds__(256)
prep_weights(const float* __restrict__ Wq, const float* __restrict__ Wk,
             const float* __restrict__ Wv, const float* __restrict__ Wo,
             unsigned short* __restrict__ Wqkv_t,
             unsigned short* __restrict__ Wo_t) {
  int idx = blockIdx.x * blockDim.x + threadIdx.x;
  const int NQKV = QKV_OUT * IN_DIM;
  if (idx < NQKV) {
    int o = idx / IN_DIM;
    int i = idx - o * IN_DIM;
    int d = o & 63;
    float v;
    if (o < 256) {
      int kv = o >> 6;
      const float* base = Wq + (size_t)i * (16 * HEAD_DIM) + (kv * GROUPS) * HEAD_DIM + d;
      v = base[0] + base[64] + base[128] + base[192];
    } else if (o < 512) {
      int h = (o - 256) >> 6;
      v = Wk[(size_t)i * O_DIM + h * HEAD_DIM + d];
    } else {
      int h = (o - 512) >> 6;
      v = Wv[(size_t)i * O_DIM + h * HEAD_DIM + d];
    }
    Wqkv_t[(size_t)o * IN_DIM + i] = f2bf(v);
  } else {
    int j = idx - NQKV;                 // 0 .. IN_DIM*O_DIM-1
    int o = j / O_DIM;                  // 0..1023  (final-output feature)
    int c = j - o * O_DIM;              // 0..255   (contraction index)
    Wo_t[(size_t)o * O_DIM + c] = f2bf(Wo[(size_t)c * IN_DIM + o]);
  }
}

// ------------------------------------------------------------------
// Kernel B: QKV projection (WMMA, K=1024) + RoPE on Q/K.
// Block = 256 thr (8 waves); wave = 16(seq) x 64(out) tile; the 64x32
// weight tile is shared by all 8 waves and double-buffer staged into LDS
// via async load-to-LDS (wait -> barrier -> compute -> barrier -> prefetch).
// Q -> Qb[b][h][s][d], K -> Kb[b][h][s][d], V -> Vt[b][h][d][s] (transposed)
// ------------------------------------------------------------------
__global__ void __launch_bounds__(256)
qkv_rope(const float* __restrict__ X, const unsigned short* __restrict__ Wqkv_t,
         unsigned short* __restrict__ Qb, unsigned short* __restrict__ Kb,
         unsigned short* __restrict__ Vt) {
  __shared__ __attribute__((aligned(16))) unsigned short Wtile[2][64 * 32];

  const int t     = threadIdx.x;
  const int lane  = t & 31;
  const int wave  = t >> 5;
  const int row16 = lane & 15;
  const int half  = lane >> 4;
  const int b  = blockIdx.z;
  const int s0 = blockIdx.y * 128 + wave * 16;
  const int o0 = blockIdx.x * 64;

  // staging slot: thread t copies 16B of row (t>>2), chunk (t&3)
  const int srow = t >> 2, schunk = t & 3;
  const unsigned short* gW = Wqkv_t + (size_t)(o0 + srow) * IN_DIM + schunk * 8;
  unsigned short* lW0 = &Wtile[0][srow * 32 + schunk * 8];
  unsigned short* lW1 = &Wtile[1][srow * 32 + schunk * 8];

  stage16(gW + 0,  lW0);   // k=0  -> buf0
  stage16(gW + 32, lW1);   // k=32 -> buf1

  const float* Xrow = X + (size_t)(b * SEQ + s0 + row16) * IN_DIM;
  v8f acc[4] = {};

  for (int k0 = 0; k0 < IN_DIM; k0 += 32) {
    const int cur = (k0 >> 5) & 1;
#if HAVE_ASYNC_LDS
    if (k0 + 32 < IN_DIM) __asm__ volatile("s_wait_asynccnt 1" ::: "memory");
    else                  __asm__ volatile("s_wait_asynccnt 0" ::: "memory");
#endif
    __syncthreads();                       // publish current tile to all waves

    __builtin_prefetch(Xrow + k0 + 64, 0, 0);   // global_prefetch_b8

    // A fragment from X (fp32 -> bf16)
    FragBF a;
    const float4* p0 = (const float4*)(Xrow + k0 + half * 8);
    const float4* p1 = (const float4*)(Xrow + k0 + 16 + half * 8);
    float4 f0 = p0[0], f1 = p0[1], f2 = p1[0], f3 = p1[1];
    a.h[0]=f2bf(f0.x);  a.h[1]=f2bf(f0.y);  a.h[2]=f2bf(f0.z);  a.h[3]=f2bf(f0.w);
    a.h[4]=f2bf(f1.x);  a.h[5]=f2bf(f1.y);  a.h[6]=f2bf(f1.z);  a.h[7]=f2bf(f1.w);
    a.h[8]=f2bf(f2.x);  a.h[9]=f2bf(f2.y);  a.h[10]=f2bf(f2.z); a.h[11]=f2bf(f2.w);
    a.h[12]=f2bf(f3.x); a.h[13]=f2bf(f3.y); a.h[14]=f2bf(f3.z); a.h[15]=f2bf(f3.w);

    // B fragments from the shared LDS tile
#pragma unroll
    for (int nt = 0; nt < 4; ++nt) {
      FragBF bb;
      const unsigned short* wr = &Wtile[cur][(nt * 16 + row16) * 32 + half * 8];
      bb.q[0] = *(const uint4*)(wr);
      bb.q[1] = *(const uint4*)(wr + 16);
      acc[nt] = wmma_bf16(a.v, bb.v, acc[nt]);
    }

    __syncthreads();                       // all waves done reading 'cur'
    if (k0 + 64 < IN_DIM)                  // prefetch 2 steps ahead into 'cur'
      stage16(gW + k0 + 64, cur ? lW1 : lW0);
  }

  // RoPE (Q/K regions only) + scatter stores.
#pragma unroll
  for (int nt = 0; nt < 4; ++nt) {
    const int o = o0 + nt * 16 + row16;   // output feature
    const int d = o & 63;
    const bool isV = (o >= 512);
#pragma unroll
    for (int r = 0; r < 8; ++r) {
      const int seq = s0 + half * 8 + r;
      float x = acc[nt][r];
      float y = __shfl_xor(x, 1);         // partner feature (d^1), same row
      float val;
      if (!isV) {
        int jj = d >> 1;
        float theta = __expf(-(float)jj * 2.302585093f);  // 10^-j
        float ang = (float)(seq + 1) * theta;
        float cc = __cosf(ang), ss = __sinf(ang);
        val = (d & 1) ? (y * ss + x * cc) : (x * cc - y * ss);
      } else {
        val = x;
      }
      unsigned short bf = f2bf(val);
      if (o < 256) {
        Qb[((size_t)(b * KV_HEADS + (o >> 6)) * SEQ + seq) * HEAD_DIM + d] = bf;
      } else if (o < 512) {
        int o2 = o - 256;
        Kb[((size_t)(b * KV_HEADS + (o2 >> 6)) * SEQ + seq) * HEAD_DIM + d] = bf;
      } else {
        int o2 = o - 512;
        Vt[((size_t)(b * KV_HEADS + (o2 >> 6)) * HEAD_DIM + d) * SEQ + seq] = bf;
      }
    }
  }
}

// ------------------------------------------------------------------
// Kernel C: causal flash attention. 8 waves/block; each wave owns a
// 16-query tile. Online softmax; P round-trips through a per-wave LDS
// slice to become an A fragment (DS ops are in-order per wave).
// ------------------------------------------------------------------
__global__ void __launch_bounds__(256)
flash_attn(const unsigned short* __restrict__ Qb,
           const unsigned short* __restrict__ Kb,
           const unsigned short* __restrict__ Vt,
           unsigned short* __restrict__ AO) {
  __shared__ __attribute__((aligned(16))) unsigned short Plds[8][16 * 32];
  const int lane  = threadIdx.x & 31;
  const int wave  = threadIdx.x >> 5;
  const int row16 = lane & 15;
  const int half  = lane >> 4;
  const int s0 = blockIdx.x * 128 + wave * 16;
  const int h  = blockIdx.y;
  const int b  = blockIdx.z;
  const size_t bh = (size_t)(b * KV_HEADS + h);
  unsigned short* myP = Plds[wave];

  // Persistent Q fragments (16 x 64 = two k-chunks)
  FragBF qa[2];
  {
    const unsigned short* Qrow = Qb + (bh * SEQ + s0 + row16) * HEAD_DIM;
#pragma unroll
    for (int kk = 0; kk < 2; ++kk) {
      const unsigned short* p = Qrow + kk * 32 + half * 8;
      qa[kk].q[0] = *(const uint4*)(p);
      qa[kk].q[1] = *(const uint4*)(p + 16);
    }
  }

  v8f acc[4] = {};
  float m[8], lsum[8];
#pragma unroll
  for (int r = 0; r < 8; ++r) { m[r] = -1e30f; lsum[r] = 0.f; }

  const int nkb = (s0 + 16 + 31) >> 5;      // causal: key blocks of 32
  for (int kb = 0; kb < nkb; ++kb) {
    const int a0 = kb * 32;
    v8f sc[2] = {};
#pragma unroll
    for (int nt = 0; nt < 2; ++nt) {
      const unsigned short* Krow =
          Kb + (bh * SEQ + a0 + nt * 16 + row16) * HEAD_DIM;
#pragma unroll
      for (int kk = 0; kk < 2; ++kk) {
        FragBF bb;
        const unsigned short* p = Krow + kk * 32 + half * 8;
        bb.q[0] = *(const uint4*)(p);
        bb.q[1] = *(const uint4*)(p + 16);
        sc[nt] = wmma_bf16(qa[kk].v, bb.v, sc[nt]);
      }
    }

    // scale, causal mask, online softmax bookkeeping
    float p0[8], p1[8], alpha[8];
#pragma unroll
    for (int r = 0; r < 8; ++r) {
      const int seq = s0 + half * 8 + r;
      float z0 = sc[0][r] * 0.125f;
      float z1 = sc[1][r] * 0.125f;
      if (a0 + row16 > seq)      z0 = -1e30f;
      if (a0 + 16 + row16 > seq) z1 = -1e30f;
      float mx = fmaxf(z0, z1);
      mx = fmaxf(mx, __shfl_xor(mx, 1));
      mx = fmaxf(mx, __shfl_xor(mx, 2));
      mx = fmaxf(mx, __shfl_xor(mx, 4));
      mx = fmaxf(mx, __shfl_xor(mx, 8));
      float mnew = fmaxf(m[r], mx);
      float al = __expf(m[r] - mnew);
      float e0 = __expf(z0 - mnew);
      float e1 = __expf(z1 - mnew);
      float rs = e0 + e1;
      rs += __shfl_xor(rs, 1);
      rs += __shfl_xor(rs, 2);
      rs += __shfl_xor(rs, 4);
      rs += __shfl_xor(rs, 8);
      lsum[r] = lsum[r] * al + rs;
      m[r] = mnew;
      alpha[r] = al;
      p0[r] = e0;
      p1[r] = e1;
    }
#pragma unroll
    for (int nt = 0; nt < 4; ++nt)
#pragma unroll
      for (int r = 0; r < 8; ++r) acc[nt][r] *= alpha[r];

    // P (C-layout) -> LDS row-major 16x32 bf16 (per-wave slice)
#pragma unroll
    for (int r = 0; r < 8; ++r) {
      const int rr = half * 8 + r;
      myP[rr * 32 + row16]      = f2bf(p0[r]);
      myP[rr * 32 + 16 + row16] = f2bf(p1[r]);
    }
    __asm__ volatile("s_wait_dscnt 0" ::: "memory");  // DS in-order per wave

    // reload P as an A fragment (16 x 32)
    FragBF pa;
    {
      const unsigned short* pr = myP + row16 * 32 + half * 8;
      pa.q[0] = *(const uint4*)(pr);
      pa.q[1] = *(const uint4*)(pr + 16);
    }

    // P @ V  (V stored transposed: Vt[d][s])
#pragma unroll
    for (int nt = 0; nt < 4; ++nt) {
      const unsigned short* Vrow =
          Vt + (bh * HEAD_DIM + nt * 16 + row16) * SEQ + a0 + half * 8;
      FragBF bb;
      bb.q[0] = *(const uint4*)(Vrow);
      bb.q[1] = *(const uint4*)(Vrow + 16);
      acc[nt] = wmma_bf16(pa.v, bb.v, acc[nt]);
    }
  }

  // finalize: divide by row sums, store AO[b][s][h*64+d] (bf16)
#pragma unroll
  for (int r = 0; r < 8; ++r) {
    const float inv = 1.0f / lsum[r];
    const int seq = s0 + half * 8 + r;
#pragma unroll
    for (int nt = 0; nt < 4; ++nt) {
      AO[(size_t)(b * SEQ + seq) * O_DIM + h * HEAD_DIM + nt * 16 + row16] =
          f2bf(acc[nt][r] * inv);
    }
  }
}

// ------------------------------------------------------------------
// Kernel D: output projection AO[4,2048,256] @ Wo[256,1024] -> fp32 out.
// ------------------------------------------------------------------
__global__ void __launch_bounds__(256)
out_proj(const unsigned short* __restrict__ AO,
         const unsigned short* __restrict__ Wo_t,
         float* __restrict__ out) {
  const int lane  = threadIdx.x & 31;
  const int wave  = threadIdx.x >> 5;
  const int row16 = lane & 15;
  const int half  = lane >> 4;
  const int b  = blockIdx.z;
  const int s0 = blockIdx.y * 128 + wave * 16;
  const int o0 = blockIdx.x * 64;

  const unsigned short* Arow = AO + (size_t)(b * SEQ + s0 + row16) * O_DIM;
  v8f acc[4] = {};
  for (int k0 = 0; k0 < O_DIM; k0 += 32) {
    FragBF a;
    const unsigned short* p = Arow + k0 + half * 8;
    a.q[0] = *(const uint4*)(p);
    a.q[1] = *(const uint4*)(p + 16);
#pragma unroll
    for (int nt = 0; nt < 4; ++nt) {
      FragBF bb;
      const unsigned short* Wrow =
          Wo_t + (size_t)(o0 + nt * 16 + row16) * O_DIM + k0 + half * 8;
      bb.q[0] = *(const uint4*)(Wrow);
      bb.q[1] = *(const uint4*)(Wrow + 16);
      acc[nt] = wmma_bf16(a.v, bb.v, acc[nt]);
    }
  }
#pragma unroll
  for (int nt = 0; nt < 4; ++nt)
#pragma unroll
    for (int r = 0; r < 8; ++r) {
      const int seq = s0 + half * 8 + r;
      out[(size_t)(b * SEQ + seq) * IN_DIM + o0 + nt * 16 + row16] = acc[nt][r];
    }
}

// ------------------------------------------------------------------
extern "C" void kernel_launch(void* const* d_in, const int* in_sizes, int n_in,
                              void* d_out, int out_size, void* d_ws, size_t ws_size,
                              hipStream_t stream) {
  const float* q  = (const float*)d_in[0];
  // d_in[1] = mask (causal; recomputed analytically)
  const float* Wq = (const float*)d_in[2];
  const float* Wk = (const float*)d_in[3];
  const float* Wv = (const float*)d_in[4];
  const float* Wo = (const float*)d_in[5];

  unsigned short* Wqkv_t = (unsigned short*)d_ws;                       // 768*1024
  unsigned short* Wo_t   = Wqkv_t + (size_t)QKV_OUT * IN_DIM;           // 1024*256
  unsigned short* Qb     = Wo_t + (size_t)IN_DIM * O_DIM;               // 4*4*2048*64
  unsigned short* Kb     = Qb + (size_t)BATCH * KV_HEADS * SEQ * HEAD_DIM;
  unsigned short* Vt     = Kb + (size_t)BATCH * KV_HEADS * SEQ * HEAD_DIM;
  unsigned short* AO     = Vt + (size_t)BATCH * KV_HEADS * SEQ * HEAD_DIM;

  prep_weights<<<4096, 256, 0, stream>>>(Wq, Wk, Wv, Wo, Wqkv_t, Wo_t);
  qkv_rope<<<dim3(QKV_OUT / 64, SEQ / 128, BATCH), 256, 0, stream>>>(q, Wqkv_t, Qb, Kb, Vt);
  flash_attn<<<dim3(SEQ / 128, KV_HEADS, BATCH), 256, 0, stream>>>(Qb, Kb, Vt, AO);
  out_proj<<<dim3(IN_DIM / 64, SEQ / 128, BATCH), 256, 0, stream>>>(AO, Wo_t, (float*)d_out);
}
